// BiLSTM_7945689497983
// MI455X (gfx1250) — matrix operand order repaired and checked by reading, not compiled, not run on hardware
//
#include <hip/hip_runtime.h>
#include <hip/hip_bf16.h>
#include <hip/hip_fp16.h>

// BiLSTM tagger for MI455X (gfx1250, wave32, WMMA).
//   B=64, S=256, E=256, H2=256, gates=1024, TAGS=20
// Pipeline (all on `stream`):
//   k_init    : zero sync flags + h0 double-buffer (graph-replay safe)
//   k_weights : transpose+convert Wx/Wh (both dirs) to f16 [N][K] for WMMA-B
//   k_gather  : embedding gather -> time-major f16 A matrix [S*B][E]
//   k_xproj   : WMMA GEMM  xproj[dir] = Aemb @ WxT^T + b  (f32, NT-streamed)
//   k_recur   : 16 persistent WGs (2 dirs x 8 column-eighths). Each WG caches
//               its 64KB Wh^T slice in LDS ONCE (lane-swizzled, conflict-free
//               ds_load_b128), then runs 256 serial steps of
//               [64,256]x[256,1024] via v_wmma_f32_16x16x32_f16 with gate
//               fusion + cell state in VGPRs; h exchanged through L2 with
//               double buffering + per-step arrival counters.
//   k_tags    : h @ W_tag + b_tag with W_tag cached in LDS (40KB)
// L2 strategy: Wh lives in LDS (zero repeated traffic on the serial path);
// xproj/h_all are single-use streams -> non-temporal hints keep them from
// polluting the 192MB L2 where the h ping-pong buffers stay hot.
// Workspace use ~162 MB (dominated by f32 xproj), assumed available.

#define S_LEN 256
#define BATCH 64
#define EMBED 256
#define H2V   256
#define G4    1024
#define HID   512
#define NTAGS 20

typedef __attribute__((ext_vector_type(16))) _Float16 v16h;
typedef __attribute__((ext_vector_type(8)))  _Float16 v8h;
typedef __attribute__((ext_vector_type(8)))  float    v8f;

__device__ __forceinline__ v8f wmma_f16(v16h a, v16h b, v8f c) {
  // D = A(16x32 f16) * B(32x16 f16) + C(16x16 f32)
  return __builtin_amdgcn_wmma_f32_16x16x32_f16(false, a, false, b, (short)0, c,
                                                false, false);
}

// A operand: 16x32 f16 tile from row-major [rows][ldk].
// lanes 0-15: M=m0+lane, K=k0..k0+15
// lanes16-31: M=m0+lane-16, K=k0+16..k0+31
__device__ __forceinline__ v16h load_a16(const _Float16* base, int ldk, int m0,
                                         int k0, int lane) {
  const int half8 = (lane >> 4) * 16;
  const _Float16* p = base + (size_t)(m0 + (lane & 15)) * ldk + k0 + half8;
  v8h lo = *(const v8h*)(p);
  v8h hi = *(const v8h*)(p + 8);
  v16h a;
#pragma unroll
  for (int i = 0; i < 8; ++i) { a[i] = lo[i]; a[i + 8] = hi[i]; }
  return a;
}

// B operand from TRANSPOSED weights [N][K] row-major in global memory.
// lanes 0-15: N=n0+lane, K=k0..k0+15 ; lanes16-31: N=n0+lane-16, K=k0+16..k0+31
__device__ __forceinline__ v16h load_b16(const _Float16* baseT, int ldk, int n0,
                                         int k0, int lane) {
  const int half16 = (lane >> 4) * 16;
  const _Float16* p = baseT + (size_t)(n0 + (lane & 15)) * ldk + k0 + half16;
  v8h lo = *(const v8h*)(p);
  v8h hi = *(const v8h*)(p + 8);
  v16h b;
#pragma unroll
  for (int i = 0; i < 8; ++i) { b[i] = lo[i]; b[i + 8] = hi[i]; }
  return b;
}

// B operand from a lane-swizzled LDS entry (512 halfs):
//   [lane*8 .. +7]       = lane's low  16B chunk
//   [256 + lane*8 .. +7] = lane's high 16B chunk
// Consecutive lanes hit strictly consecutive LDS addresses -> conflict-free.
__device__ __forceinline__ v16h load_b_lds(const _Float16* entry, int lane) {
  v8h lo = *(const v8h*)(entry + lane * 8);
  v8h hi = *(const v8h*)(entry + 256 + lane * 8);
  v16h b;
#pragma unroll
  for (int i = 0; i < 8; ++i) { b[i] = lo[i]; b[i + 8] = hi[i]; }
  return b;
}

__device__ __forceinline__ float sigm(float x) {
  return 1.0f / (1.0f + __expf(-x));
}

// ---------------------------------------------------------------- k_init ----
__global__ void k_init(int* flags, unsigned* hbuf_u) {
  const int t = blockIdx.x * blockDim.x + threadIdx.x;
  const int stride = gridDim.x * blockDim.x;
  for (int i = t; i < 2 * S_LEN; i += stride) flags[i] = 0;
  // h double buffers: 2 dirs * 2 bufs * 64 * 256 f16 = 32768 dwords
  for (int i = t; i < (2 * 2 * BATCH * H2V) / 2; i += stride) hbuf_u[i] = 0u;
}

// ------------------------------------------------------------- k_weights ----
// Transpose [K=256][N=1024] f32 -> [N=1024][K=256] f16 (WMMA-B friendly).
__global__ void k_weights(const float* Wx_f, const float* Wx_b,
                          const float* Wh_f, const float* Wh_b,
                          _Float16* WxT, _Float16* WhT) {
  const int m = blockIdx.y;  // 0:Wx_f 1:Wx_b 2:Wh_f 3:Wh_b
  const int idx = blockIdx.x * blockDim.x + threadIdx.x;  // n*256+k
  const int n = idx >> 8, k = idx & 255;
  const float* src = (m == 0) ? Wx_f : (m == 1) ? Wx_b : (m == 2) ? Wh_f : Wh_b;
  _Float16* dst = (m == 0) ? WxT
                : (m == 1) ? (WxT + (size_t)G4 * EMBED)
                : (m == 2) ? WhT
                           : (WhT + (size_t)G4 * H2V);
  dst[idx] = (_Float16)src[(size_t)k * G4 + n];
}

// -------------------------------------------------------------- k_gather ----
// Time-major embedding matrix: Aemb[s*64+b][e] = emb[x[b][s]][e] (f16).
__global__ void k_gather(const int* x, const float* emb, _Float16* Aemb) {
  const int r = blockIdx.x;            // s*64 + b
  const int s = r >> 6, b = r & 63;
  const int tok = x[b * S_LEN + s];
  const int e = threadIdx.x;
  Aemb[(size_t)r * EMBED + e] = (_Float16)emb[(size_t)tok * EMBED + e];
}

// --------------------------------------------------------------- k_xproj ----
// xproj[dir][S*B][1024] = Aemb @ Wx^T(+bias). 8 waves/block, 1 tile/wave.
__global__ void k_xproj(const _Float16* Aemb, const _Float16* WxT,
                        const float* b_f, const float* b_b, float* xproj) {
  const int lane = threadIdx.x & 31, w = threadIdx.x >> 5;
  const int dir = blockIdx.z;
  const int m0 = blockIdx.x * 16;              // 0..16368
  const int n0 = (blockIdx.y * 8 + w) * 16;    // 0..1008
  const _Float16* BT = WxT + (size_t)dir * G4 * EMBED;
  const float* bias = dir ? b_b : b_f;
  float* C = xproj + (size_t)dir * 16384ull * G4;

  v8f acc;
#pragma unroll
  for (int r = 0; r < 8; ++r) acc[r] = 0.0f;

#pragma unroll
  for (int kk = 0; kk < 8; ++kk) {             // K = 256
    v16h a = load_a16(Aemb, EMBED, m0, kk * 32, lane);
    v16h b = load_b16(BT, EMBED, n0, kk * 32, lane);
    acc = wmma_f16(a, b, acc);
  }

  const float bv = bias[n0 + (lane & 15)];
  const int mbase = m0 + ((lane >> 4) * 8);
  const int n = n0 + (lane & 15);
#pragma unroll
  for (int r = 0; r < 8; ++r)  // single-use stream -> NT store (TH=NT)
    __builtin_nontemporal_store(acc[r] + bv, &C[(size_t)(mbase + r) * G4 + n]);
}

// --------------------------------------------------------------- k_recur ----
// 16 blocks: blockIdx.x = dir*8 + eighth. 256 threads = 8 waves; each wave
// owns one 16x16 h-tile (and its 4 gate tiles) for all 256 steps; cell state
// c stays in VGPRs; this WG's Wh^T slice (128 gate rows) lives in 64KB LDS.
__global__ void __launch_bounds__(256) k_recur(const float* xproj,
                                               const _Float16* WhT,
                                               _Float16* hbuf,
                                               _Float16* h_all, int* flags) {
  // 4 gates * 2 j-tiles * 8 k-steps, each entry = 32 lanes * 32B = 1KB.
  __shared__ _Float16 sWh[4 * 2 * 8 * 512];  // 65536 B
  const int tid = threadIdx.x, lane = tid & 31, w = tid >> 5;
  const int dir = (int)blockIdx.x >> 3, q = (int)blockIdx.x & 7;
  const int mtile = w & 3, jt = w >> 2;         // 4 M-tiles x 2 j-tiles
  const int m0 = mtile * 16;                    // batch-tile
  const int j0 = q * 32 + jt * 16;              // h-column tile (0..240)
  const _Float16* Wt = WhT + (size_t)dir * G4 * H2V;
  const float* xp_dir = xproj + (size_t)dir * 16384ull * G4;
  _Float16* hb = hbuf + (size_t)dir * 2 * BATCH * H2V;
  int* flg = flags + dir * S_LEN;

  // ---- one-time fill of the LDS weight cache (lane-swizzled entries) ----
  // entry e = g*16 + jtile*8 + kk ; rows n = g*256 + q*32 + jtile*16 + lane%16
#pragma unroll
  for (int i = 0; i < 8; ++i) {
    const int e = w * 8 + i;                    // 64 entries over 8 waves
    const int g = e >> 4, ejt = (e >> 3) & 1, kk = e & 7;
    const int n0 = g * H2V + q * 32 + ejt * 16;
    const int half16 = (lane >> 4) * 16;
    const _Float16* p =
        Wt + (size_t)(n0 + (lane & 15)) * H2V + kk * 32 + half16;
    _Float16* dstE = sWh + e * 512;
    *(v8h*)(dstE + lane * 8) = *(const v8h*)(p);
    *(v8h*)(dstE + 256 + lane * 8) = *(const v8h*)(p + 8);
  }
  __syncthreads();

  const int ncol = lane & 15;
  const int mbase = m0 + ((lane >> 4) * 8);

  float cst[8];
#pragma unroll
  for (int r = 0; r < 8; ++r) cst[r] = 0.0f;

  for (int iter = 0; iter < S_LEN; ++iter) {
    const int t = dir ? (S_LEN - 1 - iter) : iter;
    const float* xp = xp_dir + (size_t)t * BATCH * G4;

    // C init = precomputed input projection (f32, NT-streamed).
    v8f acc[4];
#pragma unroll
    for (int g = 0; g < 4; ++g) {
      const int n = g * H2V + j0 + ncol;
#pragma unroll
      for (int r = 0; r < 8; ++r)
        acc[g][r] =
            __builtin_nontemporal_load(&xp[(size_t)(mbase + r) * G4 + n]);
    }
    if (iter + 1 < S_LEN) {  // prefetch next step's activations
      const float* xpn = xp_dir + (size_t)(dir ? t - 1 : t + 1) * BATCH * G4;
      __builtin_prefetch(&xpn[(size_t)mbase * G4 + j0 + ncol], 0, 1);
    }

    const _Float16* hprev = hb + (size_t)(iter & 1) * BATCH * H2V;
    _Float16* hnext = hb + (size_t)((iter + 1) & 1) * BATCH * H2V;

    // gates += h_prev @ Wh : A tile from L2-hot h, B tiles from LDS.
#pragma unroll
    for (int kk = 0; kk < 8; ++kk) {
      v16h a = load_a16(hprev, H2V, m0, kk * 32, lane);
#pragma unroll
      for (int g = 0; g < 4; ++g) {
        v16h b = load_b_lds(sWh + (g * 16 + jt * 8 + kk) * 512, lane);
        acc[g] = wmma_f16(a, b, acc[g]);
      }
    }

    // fused i/f/g/o -> c,h (f32 in VGPRs)
#pragma unroll
    for (int r = 0; r < 8; ++r) {
      const float iv = sigm(acc[0][r]);
      const float fv = sigm(acc[1][r]);
      const float gv = tanhf(acc[2][r]);
      const float ov = sigm(acc[3][r]);
      const float cn = fv * cst[r] + iv * gv;
      cst[r] = cn;
      const float hv = ov * tanhf(cn);
      const int m = mbase + r;      // batch index
      const int j = j0 + ncol;      // h column
      hnext[(size_t)m * H2V + j] = (_Float16)hv;
      __builtin_nontemporal_store(
          (_Float16)hv, &h_all[((size_t)m * S_LEN + t) * HID + dir * H2V + j]);
    }

    // publish our h eighth, then wait for all 8 eighths of this direction
    __threadfence();
    __syncthreads();
    if (tid == 0) {
      atomicAdd(&flg[iter], 1);
      while (__hip_atomic_load(&flg[iter], __ATOMIC_ACQUIRE,
                               __HIP_MEMORY_SCOPE_AGENT) < 8)
        __builtin_amdgcn_s_sleep(1);
    }
    __syncthreads();
    __threadfence();  // acquire: invalidate so next-step h loads are fresh
  }
}

// ---------------------------------------------------------------- k_tags ----
// out[row][tag] = h_all[row] . W_tag[:,tag] + b_tag ; W_tag cached in LDS.
__global__ void k_tags(const _Float16* h_all, const float* W_tag,
                       const float* b_tag, float* out) {
  __shared__ float sW[HID * NTAGS];   // 40 KB
  __shared__ float sb[NTAGS];
  for (int i = threadIdx.x; i < HID * NTAGS; i += blockDim.x) sW[i] = W_tag[i];
  if (threadIdx.x < NTAGS) sb[threadIdx.x] = b_tag[threadIdx.x];
  __syncthreads();

  const int row = blockIdx.x * blockDim.x + threadIdx.x;  // b*256 + s
  const _Float16* h = h_all + (size_t)row * HID;
  float acc[NTAGS];
#pragma unroll
  for (int t = 0; t < NTAGS; ++t) acc[t] = sb[t];
  for (int k = 0; k < HID; k += 8) {
    v8h hv8 = __builtin_nontemporal_load((const v8h*)(h + k));
#pragma unroll
    for (int j = 0; j < 8; ++j) {
      const float hv = (float)hv8[j];
      const float* wrow = &sW[(k + j) * NTAGS];
#pragma unroll
      for (int t = 0; t < NTAGS; ++t) acc[t] += hv * wrow[t];
    }
  }
  float* o = out + (size_t)row * NTAGS;
#pragma unroll
  for (int t = 0; t < NTAGS; ++t) o[t] = acc[t];
}

// ---------------------------------------------------------------- launch ----
extern "C" void kernel_launch(void* const* d_in, const int* in_sizes, int n_in,
                              void* d_out, int out_size, void* d_ws,
                              size_t ws_size, hipStream_t stream) {
  (void)in_sizes; (void)n_in; (void)out_size; (void)ws_size;
  const int*   x     = (const int*)d_in[0];
  const float* emb   = (const float*)d_in[1];
  const float* Wx_f  = (const float*)d_in[2];
  const float* Wh_f  = (const float*)d_in[3];
  const float* b_f   = (const float*)d_in[4];
  const float* Wx_b  = (const float*)d_in[5];
  const float* Wh_b  = (const float*)d_in[6];
  const float* b_b   = (const float*)d_in[7];
  const float* W_tag = (const float*)d_in[8];
  const float* b_tag = (const float*)d_in[9];
  float* out = (float*)d_out;

  char* ws = (char*)d_ws;
  size_t off = 0;
  auto alloc = [&](size_t bytes) {
    char* p = ws + off;
    off += (bytes + 255) & ~(size_t)255;
    return p;
  };
  _Float16* Aemb  = (_Float16*)alloc(16384ull * EMBED * 2);     //   8 MB
  _Float16* WxT   = (_Float16*)alloc(2ull * G4 * EMBED * 2);    //   1 MB
  _Float16* WhT   = (_Float16*)alloc(2ull * G4 * H2V * 2);      //   1 MB
  float*    xproj = (float*)   alloc(2ull * 16384ull * G4 * 4); // 128 MB
  _Float16* hbuf  = (_Float16*)alloc(2ull * 2 * BATCH * H2V * 2);
  _Float16* h_all = (_Float16*)alloc(16384ull * HID * 2);       //  16 MB
  int*      flags = (int*)     alloc(2ull * S_LEN * 4);

  k_init<<<64, 256, 0, stream>>>(flags, (unsigned*)hbuf);
  k_weights<<<dim3(1024, 4), 256, 0, stream>>>(Wx_f, Wx_b, Wh_f, Wh_b, WxT, WhT);
  k_gather<<<16384, 256, 0, stream>>>(x, emb, Aemb);
  k_xproj<<<dim3(1024, 8, 2), 256, 0, stream>>>(Aemb, WxT, b_f, b_b, xproj);
  k_recur<<<16, 256, 0, stream>>>(xproj, WhT, hbuf, h_all, flags);
  k_tags<<<64, 256, 0, stream>>>(h_all, W_tag, b_tag, out);
}
